// TextLSTM_51161650430708
// MI455X (gfx1250) — compile-verified, matrix operand order checked by reading, not run on hardware
//
#include <hip/hip_runtime.h>
#include <stdint.h>

#define SEQ   2048
#define BATCH 128
#define HID   128
#define NCLS  50257

typedef __attribute__((ext_vector_type(16))) __bf16   v16bf;
typedef __attribute__((ext_vector_type(8)))  float    v8f;
typedef __attribute__((ext_vector_type(4)))  uint32_t v4u;

union Frag {
  v16bf    v;
  uint32_t u[8];
  v4u      q[2];
};

__device__ __forceinline__ uint32_t bf16_rne(float x) {
  uint32_t u = __float_as_uint(x);
  u += 0x7FFFu + ((u >> 16) & 1u);
  return u >> 16;
}
__device__ __forceinline__ uint32_t pack2_bf16(float lo, float hi) {
  return bf16_rne(lo) | (bf16_rne(hi) << 16);
}

__device__ __forceinline__ float sigm(float x) {
  return 1.0f / (1.0f + __expf(-x));
}
__device__ __forceinline__ float tanh_fast(float x) {
  float e = __expf(2.0f * x);
  return (e - 1.0f) / (e + 1.0f);
}

// A-fragment (16x32 bf16) gathered from an f32 row (stride-1 K)
__device__ __forceinline__ v16bf gather_frag_f32(const float* rowp, int kb, int khalf) {
  Frag f;
  const int k0 = kb * 32 + khalf * 8;
#pragma unroll
  for (int d = 0; d < 8; ++d) {
    int k = k0 + ((d & 4) << 2) + (d & 3) * 2;
    f.u[d] = pack2_bf16(rowp[k], rowp[k + 1]);
  }
  return f.v;
}

// fragment load at (base + literal offset): two 16B halves
__device__ __forceinline__ v16bf load_frag_at(const char* p) {
  Frag f;
  f.q[0] = *(const v4u*)p;
  f.q[1] = *(const v4u*)(p + 16);
  return f.v;
}

__device__ __forceinline__ v8f wmma_bf16(v16bf a, v16bf b, v8f c) {
  return __builtin_amdgcn_wmma_f32_16x16x32_bf16(false, a, false, b, (short)0, c, false, false);
}

// unpack 8 packed bf16 (16B) -> 8 floats
__device__ __forceinline__ void unpack8_bf16(const char* p, float* out8) {
  v4u q = *(const v4u*)p;
#pragma unroll
  for (int d = 0; d < 4; ++d) {
    out8[2 * d]     = __uint_as_float(q[d] << 16);
    out8[2 * d + 1] = __uint_as_float(q[d] & 0xFFFF0000u);
  }
}

// byte address of element (rt = row/16, mp = row%16, col) inside an
// A-fragment-layout h buffer: [rt][kb][lane][8 dwords]  (cold paths only)
__device__ __forceinline__ char* h_elem_addr(char* hbuf, int rt, int mp, int col) {
  int kb2  = col >> 5;
  int kc   = col & 31;
  int grp  = (kc >> 3) & 1;
  int lane = mp + 16 * grp;
  int d    = ((kc & 16) >> 2) + ((kc & 7) >> 1);
  return hbuf + ((rt * 4 + kb2) * 32 + lane) * 32 + d * 4 + (kc & 1) * 2;
}

// ---------------------------------------------------------------------------
// Kernel 1: pack weights (Wh, Wi1, Wi2 -> B-fragment blobs) and biases
// (bf/bi/bg/bo -> accumulator-layout bf16 blob). 20480 threads.
// ---------------------------------------------------------------------------
struct P16 { const float* p[16]; };

__global__ void pack_params(P16 W, uint32_t* __restrict__ ws) {
  int tid = blockIdx.x * blockDim.x + threadIdx.x;
  if (tid < 12288) {
    int mat  = tid / 4096;
    int rem  = tid % 4096;
    int c    = rem / 128;        // gate*8 + htile
    int k2   = rem % 128;
    int kb   = k2 / 32;
    int lane = k2 % 32;
    int g = c >> 3, ht = c & 7;
    const float* src = W.p[mat * 4 + g];
    int n = ht * 16 + (lane & 15);
    int khalf = lane >> 4;
    uint32_t* dst = ws + mat * 32768 + ((c * 4 + kb) * 32 + lane) * 8;
    int k0 = kb * 32 + khalf * 8;
#pragma unroll
    for (int d = 0; d < 8; ++d) {
      int k = k0 + ((d & 4) << 2) + (d & 3) * 2;
      dst[d] = pack2_bf16(src[k * 128 + n], src[(k + 1) * 128 + n]);
    }
  } else if (tid < 20480) {
    // bias blob in accumulator layout: [(w*16+g*4+u)*32+lane][4 dw]
    int t2   = tid - 12288;           // 0..8191
    int lane = t2 & 31;
    int rest = t2 >> 5;               // w*16 + g*4 + u
    int u = rest & 3;
    int g = (rest >> 2) & 3;
    int w = rest >> 4;                // 0..15
    int r = w & 7, cg = w >> 3;
    int m = lane & 15, khalf = lane >> 4;
    int rowD = 16 * r + 8 * khalf;
    int col  = cg * 64 + m + u * 16;
    const float* bp = W.p[12 + g];
    uint32_t* dst = ws + 98304 + t2 * 4;
#pragma unroll
    for (int d = 0; d < 4; ++d)
      dst[d] = pack2_bf16(bp[(rowD + 2 * d) * 128 + col],
                          bp[(rowD + 2 * d + 1) * 128 + col]);
  }
}

// ---------------------------------------------------------------------------
// Kernel 2: fused 2-layer LSTM scan. One persistent workgroup, 16 waves.
// LDS (320 KB exactly): Wh frag blob 128K + bias blob 128K + h1/h2 frag
// buffers 32K each. Wi1/Wi2 streamed from L2 with cached b128 loads; an
// asm memory clobber per step stops LICM from hoisting them into scratch.
// All hot-loop addresses are base pointer + literal offset.
// ---------------------------------------------------------------------------
#define WH_LDS   0
#define BIAS_LDS 131072
#define H1_LDS   262144
#define H2_LDS   294912

__global__ __launch_bounds__(512) void lstm_fused(
    const int*   __restrict__ X,   const float* __restrict__ h0,
    const float* __restrict__ c0,  const float* __restrict__ emb,
    const uint32_t* __restrict__ whBlob, const uint32_t* __restrict__ wi1Blob,
    const uint32_t* __restrict__ wi2Blob, const uint32_t* __restrict__ biasBlob,
    float* __restrict__ h2out) {
  extern __shared__ char lds[];
  char* whLds   = lds + WH_LDS;
  char* biasLds = lds + BIAS_LDS;
  char* h1b     = lds + H1_LDS;
  char* h2b     = lds + H2_LDS;
  const int tid = threadIdx.x;

  // stage Wh + bias blobs into LDS
  {
    uint4* dw = (uint4*)whLds;
    const uint4* sw = (const uint4*)whBlob;
    uint4* db = (uint4*)biasLds;
    const uint4* sb = (const uint4*)biasBlob;
#pragma unroll 1
    for (int i = tid; i < 8192; i += 512) {
      dw[i] = sw[i];
      db[i] = sb[i];
    }
  }
  // init h1/h2 (fragment layout) from h0 as bf16
#pragma unroll 1
  for (int e = tid; e < 2 * 128 * 64; e += 512) {
    int layer = e >> 13;
    int rem = e & 8191;
    int row = rem >> 6;
    int col = (rem & 63) * 2;
    const float* hp = h0 + layer * 16384 + row * 128 + col;
    char* buf = layer ? h2b : h1b;
    *(uint32_t*)h_elem_addr(buf, row >> 4, row & 15, col) = pack2_bf16(hp[0], hp[1]);
  }
  __syncthreads();

  const int wv = tid >> 5, lane = tid & 31;
  const int r = wv & 7, cg = wv >> 3;          // 8 row-tiles x 2 col-groups
  const int m = lane & 15, khalf = lane >> 4;
  const int rowA = 16 * r + m;                 // batch row for X/emb gather
  const int rowD = 16 * r + 8 * khalf;         // D/C row base (+vk)
  const int colD = cg * 64 + m;                // D/C col base (+u*16)

  // hot-loop base pointers (all further addressing is literal offsets)
  const char* whP  = whLds + cg * 16384 + lane * 32;   // + g*32768 + u*4096 + kb*1024
  const char* whPh = whP + 65536;                      // gates 2..3 (ds offset < 64K)
  const char* wi1P = (const char*)wi1Blob + cg * 16384 + lane * 32;
  const char* wi2P = (const char*)wi2Blob + cg * 16384 + lane * 32;
  const char* h1P  = h1b + r * 4096 + lane * 32;       // + kb*1024
  const char* h2P  = h2b + r * 4096 + lane * 32;
  const char* bP   = biasLds + wv * 8192 + lane * 16;  // + g*2048 + u*512
  // h elementwise store base: + (u>>1)*1024 + (u&1)*16 + vk*32
  const int hsOff = r * 4096 + cg * 2048 +
                    (8 * khalf + 16 * (m >> 3)) * 32 + ((m & 7) >> 1) * 4 + (m & 1) * 2;
  char* h1sb = h1b + hsOff;
  char* h2sb = h2b + hsOff;

  // initial cell state (registers for the whole scan)
  v8f c1[4], c2[4];
#pragma unroll
  for (int u = 0; u < 4; ++u)
#pragma unroll
    for (int vk = 0; vk < 8; ++vk) {
      c1[u][vk] = c0[(rowD + vk) * 128 + colD + u * 16];
      c2[u][vk] = c0[16384 + (rowD + vk) * 128 + colD + u * 16];
    }

  const v8f vzero = {0.f, 0.f, 0.f, 0.f, 0.f, 0.f, 0.f, 0.f};

#pragma unroll 1
  for (int t = 0; t < SEQ; ++t) {
    // full memory clobber: keeps weight/bias loads inside the loop (no LICM
    // hoist-to-scratch) while leaving them as normal cached loads.
    asm volatile("" ::: "memory");

    // ----------------- layer 1: gates = emb(x_t)@Wi1 + h1@Wh + b -----------
    const int idx = X[rowA * SEQ + t];
    const float* erow = emb + (size_t)idx * HID;
    __builtin_prefetch(erow, 0, 0);
    v8f acc[4][4];
#pragma unroll
    for (int g = 0; g < 4; ++g)
#pragma unroll
      for (int u = 0; u < 4; ++u) acc[g][u] = vzero;
#pragma unroll
    for (int kb = 0; kb < 4; ++kb) {
      v16bf ax = gather_frag_f32(erow, kb, khalf);
      v16bf ah = load_frag_at(h1P + kb * 1024);
#pragma unroll
      for (int g = 0; g < 4; ++g)
#pragma unroll
        for (int u = 0; u < 4; ++u) {
          const int off = g * 32768 + u * 4096 + kb * 1024;
          const char* whp = (g < 2) ? (whP + off) : (whPh + (off - 65536));
          acc[g][u] = wmma_bf16(ah, load_frag_at(whp), acc[g][u]);
          acc[g][u] = wmma_bf16(ax, load_frag_at(wi1P + off), acc[g][u]);
        }
    }
    __syncthreads();   // all h1(t-1) reads complete before overwriting
#pragma unroll
    for (int u = 0; u < 4; ++u) {
      float bF[8], bI[8], bG[8], bO[8];
      unpack8_bf16(bP + 0 * 2048 + u * 512, bF);
      unpack8_bf16(bP + 1 * 2048 + u * 512, bI);
      unpack8_bf16(bP + 2 * 2048 + u * 512, bG);
      unpack8_bf16(bP + 3 * 2048 + u * 512, bO);
#pragma unroll
      for (int vk = 0; vk < 8; ++vk) {
        float fg = sigm(acc[0][u][vk] + bF[vk]);
        float ig = sigm(acc[1][u][vk] + bI[vk]);
        float gg = tanh_fast(acc[2][u][vk] + bG[vk]);
        float og = sigm(acc[3][u][vk] + bO[vk]);
        float cn = fg * c1[u][vk] + ig * gg;
        c1[u][vk] = cn;
        float hv = og * tanh_fast(cn);
        *(uint16_t*)(h1sb + (u >> 1) * 1024 + (u & 1) * 16 + vk * 32) =
            (uint16_t)bf16_rne(hv);
      }
    }
    __syncthreads();   // h1(t) visible for layer 2

    // ----------------- layer 2: gates = h1_t@Wi2 + h2@Wh + b ---------------
#pragma unroll
    for (int g = 0; g < 4; ++g)
#pragma unroll
      for (int u = 0; u < 4; ++u) acc[g][u] = vzero;
#pragma unroll
    for (int kb = 0; kb < 4; ++kb) {
      v16bf ax = load_frag_at(h1P + kb * 1024);
      v16bf ah = load_frag_at(h2P + kb * 1024);
#pragma unroll
      for (int g = 0; g < 4; ++g)
#pragma unroll
        for (int u = 0; u < 4; ++u) {
          const int off = g * 32768 + u * 4096 + kb * 1024;
          const char* whp = (g < 2) ? (whP + off) : (whPh + (off - 65536));
          acc[g][u] = wmma_bf16(ah, load_frag_at(whp), acc[g][u]);
          acc[g][u] = wmma_bf16(ax, load_frag_at(wi2P + off), acc[g][u]);
        }
    }
    __syncthreads();   // all h2(t-1) reads complete before overwriting
#pragma unroll
    for (int u = 0; u < 4; ++u) {
      float bF[8], bI[8], bG[8], bO[8];
      unpack8_bf16(bP + 0 * 2048 + u * 512, bF);
      unpack8_bf16(bP + 1 * 2048 + u * 512, bI);
      unpack8_bf16(bP + 2 * 2048 + u * 512, bG);
      unpack8_bf16(bP + 3 * 2048 + u * 512, bO);
#pragma unroll
      for (int vk = 0; vk < 8; ++vk) {
        float fg = sigm(acc[0][u][vk] + bF[vk]);
        float ig = sigm(acc[1][u][vk] + bI[vk]);
        float gg = tanh_fast(acc[2][u][vk] + bG[vk]);
        float og = sigm(acc[3][u][vk] + bO[vk]);
        float cn = fg * c2[u][vk] + ig * gg;
        c2[u][vk] = cn;
        float hv = og * tanh_fast(cn);
        *(uint16_t*)(h2sb + (u >> 1) * 1024 + (u & 1) * 16 + vk * 32) =
            (uint16_t)bf16_rne(hv);
      }
    }
    __syncthreads();   // h2(t) visible for next step
  }

  // final h2 (timestep SEQ-1) -> workspace f32 [128][128]
#pragma unroll 1
  for (int e = tid; e < 128 * 128; e += 512) {
    int row = e >> 7, col = e & 127;
    uint32_t hb = *(const uint16_t*)h_elem_addr(h2b, row >> 4, row & 15, col);
    h2out[e] = __uint_as_float(hb << 16);
  }
}

// ---------------------------------------------------------------------------
// Kernel 3: out = h2 @ W_out + b   ([128,128] x [128,50257], f32 out)
// ---------------------------------------------------------------------------
__global__ __launch_bounds__(256) void out_gemm(
    const float* __restrict__ h2s, const float* __restrict__ Wout,
    const float* __restrict__ bvec, float* __restrict__ out) {
  const int tid = threadIdx.x, w = tid >> 5, lane = tid & 31;
  const int m = lane & 15, khalf = lane >> 4;
  const int ct = blockIdx.x;
  const int col = ct * 16 + m;
  const bool okc = col < NCLS;
  v8f acc = {0.f, 0.f, 0.f, 0.f, 0.f, 0.f, 0.f, 0.f};
#pragma unroll
  for (int kb = 0; kb < 4; ++kb) {
    v16bf a = gather_frag_f32(h2s + (16 * w + m) * 128, kb, khalf);
    Frag b;
    const int k0 = kb * 32 + khalf * 8;
#pragma unroll
    for (int d = 0; d < 8; ++d) {
      int k = k0 + ((d & 4) << 2) + (d & 3) * 2;
      float lo = okc ? Wout[(size_t)k * NCLS + col] : 0.f;
      float hi = okc ? Wout[(size_t)(k + 1) * NCLS + col] : 0.f;
      b.u[d] = pack2_bf16(lo, hi);
    }
    acc = wmma_bf16(a, b.v, acc);
  }
  if (okc) {
    float bb = bvec[col];
#pragma unroll
    for (int vk = 0; vk < 8; ++vk) {
      int row = 16 * w + 8 * khalf + vk;
      out[(size_t)row * NCLS + col] = acc[vk] + bb;
    }
  }
}

// ---------------------------------------------------------------------------
extern "C" void kernel_launch(void* const* d_in, const int* in_sizes, int n_in,
                              void* d_out, int out_size, void* d_ws, size_t ws_size,
                              hipStream_t stream) {
  (void)in_sizes; (void)n_in; (void)out_size; (void)ws_size;
  const int*   X    = (const int*)  d_in[0];
  const float* h0   = (const float*)d_in[1];
  const float* c0   = (const float*)d_in[2];
  const float* emb  = (const float*)d_in[3];
  const float* Wif  = (const float*)d_in[4];
  const float* Whf  = (const float*)d_in[5];
  const float* bf_  = (const float*)d_in[6];
  const float* Wii  = (const float*)d_in[7];
  const float* Whi  = (const float*)d_in[8];
  const float* bi_  = (const float*)d_in[9];
  const float* Wig  = (const float*)d_in[10];
  const float* Whg  = (const float*)d_in[11];
  const float* bg_  = (const float*)d_in[12];
  const float* Wio  = (const float*)d_in[13];
  const float* Who  = (const float*)d_in[14];
  const float* bo_  = (const float*)d_in[15];
  const float* Wif2 = (const float*)d_in[16];
  const float* Wii2 = (const float*)d_in[17];
  const float* Wig2 = (const float*)d_in[18];
  const float* Wio2 = (const float*)d_in[19];
  const float* Wout = (const float*)d_in[20];
  const float* bvec = (const float*)d_in[21];

  uint32_t* ws    = (uint32_t*)d_ws;
  uint32_t* whB   = ws;                       // 131072 B
  uint32_t* wi1B  = ws + 32768;               // 131072 B
  uint32_t* wi2B  = ws + 65536;               // 131072 B
  uint32_t* biasB = ws + 98304;               // 131072 B
  float*    h2ws  = (float*)(ws + 131072);    // 65536 B   (total ws: 576 KB)

  P16 P;
  P.p[0] = Whf;  P.p[1] = Whi;  P.p[2]  = Whg;  P.p[3]  = Who;
  P.p[4] = Wif;  P.p[5] = Wii;  P.p[6]  = Wig;  P.p[7]  = Wio;
  P.p[8] = Wif2; P.p[9] = Wii2; P.p[10] = Wig2; P.p[11] = Wio2;
  P.p[12] = bf_; P.p[13] = bi_; P.p[14] = bg_;  P.p[15] = bo_;

  pack_params<<<80, 256, 0, stream>>>(P, ws);
  lstm_fused<<<1, 512, 327680, stream>>>(X, h0, c0, emb,
                                         whB, wi1B, wi2B, biasB, h2ws);
  out_gemm<<<3142, 256, 0, stream>>>(h2ws, Wout, bvec, (float*)d_out);
}
